// MyAttentionHead_6588479832341
// MI455X (gfx1250) — matrix operand order repaired and checked by reading, not compiled, not run on hardware
//
#include <hip/hip_runtime.h>
#include <hip/hip_bf16.h>
#include <math.h>

#define BATCH 4
#define SEQ 2048
#define DMODEL 1024
#define DHEAD 64
#define ROWS (BATCH * SEQ) // 8192

typedef __attribute__((ext_vector_type(16))) __bf16 v16bf;
typedef __attribute__((ext_vector_type(8)))  __bf16 v8bf;
typedef __attribute__((ext_vector_type(8)))  float  v8f;

__device__ __forceinline__ v8f wmma_bf16(v16bf a, v16bf b, v8f c) {
  // D(16x16,f32) = A(16x32,bf16) * B(32x16,bf16) + C
  return __builtin_amdgcn_wmma_f32_16x16x32_bf16(
      false, a, false, b, (short)0, c, false, false);
}

// A-fragment (16x32 bf16). Lane L: row r = L%16, h = L/16.
// a[i] = rowbase[(i>>3)*16 + h*8 + (i&7)]  (two contiguous 16B chunks)
__device__ __forceinline__ v16bf load_a_frag(const __bf16* rowbase, int h) {
  v8bf lo = *(const v8bf*)(rowbase + h * 8);
  v8bf hi = *(const v8bf*)(rowbase + 16 + h * 8);
  v16bf a;
#pragma unroll
  for (int i = 0; i < 8; ++i) { a[i] = lo[i]; a[i + 8] = hi[i]; }
  return a;
}

// B-fragment (32x16 bf16). Lane L: col n = L%16, h = L/16.
// b[i] = colbase[(i>>3)*16 + h*8 + (i&7)]  (two contiguous 16B chunks; colbase = B^T row ptr)
__device__ __forceinline__ v16bf load_b_frag(const __bf16* colbase, int h) {
  v8bf lo = *(const v8bf*)(colbase + h * 8);
  v8bf hi = *(const v8bf*)(colbase + 16 + h * 8);
  v16bf b;
#pragma unroll
  for (int i = 0; i < 8; ++i) { b[i] = lo[i]; b[i + 8] = hi[i]; }
  return b;
}

// ---------------------------------------------------------------------------
// Kernel 0: convert weights to bf16. Q/K/V transposed to [64][1024] so GEMM
// B-fragments are contiguous; O kept [1024][64] (rows are the k-dim of y@O^T).
// ---------------------------------------------------------------------------
__global__ void __launch_bounds__(256) prep_weights(
    const float* __restrict__ Q, const float* __restrict__ K,
    const float* __restrict__ V, const float* __restrict__ O,
    __bf16* __restrict__ Qt, __bf16* __restrict__ Kt,
    __bf16* __restrict__ Vt, __bf16* __restrict__ Ob) {
  int tid = blockIdx.x * 256 + threadIdx.x;   // 65536 threads total
  int w   = tid >> 14;
  int idx = tid & 16383;
  int r = idx >> 6;   // d_model index
  int c = idx & 63;   // d_head index
  if (w == 0)      Qt[c * DMODEL + r] = (__bf16)Q[r * DHEAD + c];
  else if (w == 1) Kt[c * DMODEL + r] = (__bf16)K[r * DHEAD + c];
  else if (w == 2) Vt[c * DMODEL + r] = (__bf16)V[r * DHEAD + c];
  else             Ob[r * DHEAD + c] = (__bf16)O[r * DHEAD + c];
}

// ---------------------------------------------------------------------------
// Kernel 1: q = x@Q, k = x@K, v = x@V.  One wave per 16-row tile (512 waves).
// q,k stored [8192][64] bf16; v stored transposed [B][64][S] bf16 so that the
// attention P@V B-fragments are contiguous loads.
// ---------------------------------------------------------------------------
__global__ void __launch_bounds__(256) proj_qkv(
    const float* __restrict__ x,
    const __bf16* __restrict__ Qt, const __bf16* __restrict__ Kt,
    const __bf16* __restrict__ Vt,
    __bf16* __restrict__ qbuf, __bf16* __restrict__ kbuf,
    __bf16* __restrict__ vtbuf) {
  const int lane = threadIdx.x & 31;
  const int wid  = blockIdx.x * 8 + (threadIdx.x >> 5);   // 0..511
  const int rb = wid * 16;
  const int n = lane & 15, h = lane >> 4;

  v8f aq[4] = {}, ak[4] = {}, av[4] = {};
  const float* xrow = x + (size_t)(rb + n) * DMODEL;

  for (int kb = 0; kb < DMODEL; kb += 32) {
    // A fragment: 16 rows of x, 32 k-values, f32 -> bf16 on the fly
    v8f lo = *(const v8f*)(xrow + kb + h * 8);
    v8f hi = *(const v8f*)(xrow + kb + 16 + h * 8);
    v16bf a;
#pragma unroll
    for (int i = 0; i < 8; ++i) { a[i] = (__bf16)lo[i]; a[i + 8] = (__bf16)hi[i]; }
#pragma unroll
    for (int ct = 0; ct < 4; ++ct) {
      const size_t col = (size_t)(ct * 16 + n);
      aq[ct] = wmma_bf16(a, load_b_frag(Qt + col * DMODEL + kb, h), aq[ct]);
      ak[ct] = wmma_bf16(a, load_b_frag(Kt + col * DMODEL + kb, h), ak[ct]);
      av[ct] = wmma_bf16(a, load_b_frag(Vt + col * DMODEL + kb, h), av[ct]);
    }
  }
  // D layout: lane holds col n, rows m = 8*h + j across the 8 VGPRs
#pragma unroll
  for (int ct = 0; ct < 4; ++ct) {
    int col = ct * 16 + n;
#pragma unroll
    for (int j = 0; j < 8; ++j) {
      int grow = rb + 8 * h + j;
      qbuf[(size_t)grow * DHEAD + col] = (__bf16)aq[ct][j];
      kbuf[(size_t)grow * DHEAD + col] = (__bf16)ak[ct][j];
      int bb = grow >> 11, s = grow & (SEQ - 1);
      vtbuf[((size_t)bb * DHEAD + col) * SEQ + s] = (__bf16)av[ct][j];
    }
  }
}

// ---------------------------------------------------------------------------
// Kernel 2: causal flash attention, y = softmax(q k^T) @ v.  One wave per
// 16-query tile. Scores computed transposed (S^T = K_chunk @ q^T) so the D
// tiles are directly reusable, in-lane, as the A fragment of P@V.
// ---------------------------------------------------------------------------
__global__ void __launch_bounds__(256) attn_flash(
    const __bf16* __restrict__ qbuf, const __bf16* __restrict__ kbuf,
    const __bf16* __restrict__ vtbuf, __bf16* __restrict__ ybuf) {
  const int lane = threadIdx.x & 31;
  const int wid  = blockIdx.x * 8 + (threadIdx.x >> 5);   // 0..511
  const int bIdx = wid >> 7;            // batch
  const int qb   = (wid & 127) * 16;    // query tile base within sequence
  const int n = lane & 15, h = lane >> 4;

  const __bf16* qB = qbuf + (size_t)bIdx * SEQ * DHEAD;
  const __bf16* kB = kbuf + (size_t)bIdx * SEQ * DHEAD;
  const __bf16* vB = vtbuf + (size_t)bIdx * DHEAD * SEQ;

  // B fragment of S^T: q^T (64 x 16), lane = query column. Loaded once.
  const v16bf bq0 = load_b_frag(qB + (size_t)(qb + n) * DHEAD + 0, h);
  const v16bf bq1 = load_b_frag(qB + (size_t)(qb + n) * DHEAD + 32, h);

  v8f acc[4] = {};                 // y tile: lane = v-col, VGPR = query row
  float m = -INFINITY, l = 0.0f;   // per-query online softmax stats (lane=query)

  for (int kb = 0; kb <= qb + 15; kb += 32) {
    // S^T tiles: D[m=key][n=query]; st0 = keys kb..kb+15, st1 = kb+16..kb+31
    v8f st0 = {}, st1 = {};
    st0 = wmma_bf16(load_a_frag(kB + (size_t)(kb + n) * DHEAD + 0,  h), bq0, st0);
    st0 = wmma_bf16(load_a_frag(kB + (size_t)(kb + n) * DHEAD + 32, h), bq1, st0);
    st1 = wmma_bf16(load_a_frag(kB + (size_t)(kb + 16 + n) * DHEAD + 0,  h), bq0, st1);
    st1 = wmma_bf16(load_a_frag(kB + (size_t)(kb + 16 + n) * DHEAD + 32, h), bq1, st1);

    if (kb + 31 > qb) {            // diagonal chunk: causal mask (key > query)
      const int q = qb + n;
#pragma unroll
      for (int j = 0; j < 8; ++j) {
        int key0 = kb + 8 * h + j;
        if (key0 > q)      st0[j] = -INFINITY;
        if (key0 + 16 > q) st1[j] = -INFINITY;
      }
    }
    // online softmax (lane pair L / L^16 holds the same query)
    float cm = -INFINITY;
#pragma unroll
    for (int j = 0; j < 8; ++j) cm = fmaxf(cm, fmaxf(st0[j], st1[j]));
    cm = fmaxf(cm, __shfl_xor(cm, 16, 32));
    const float mn = fmaxf(m, cm);

    float ls = 0.0f;
    v16bf pa;                       // P as A fragment, built in-lane from D tiles
#pragma unroll
    for (int j = 0; j < 8; ++j) {
      float p0 = __expf(st0[j] - mn);
      float p1 = __expf(st1[j] - mn);
      ls += p0 + p1;
      pa[j]     = (__bf16)p0;
      pa[j + 8] = (__bf16)p1;
    }
    ls += __shfl_xor(ls, 16, 32);
    const float alpha = __expf(m - mn);
    l = l * alpha + ls;
    m = mn;

    // rescale accumulator: row m = 8h+j needs alpha of that query (in lane 8h+j)
    float aj[8];
#pragma unroll
    for (int j = 0; j < 8; ++j) aj[j] = __shfl(alpha, 8 * h + j, 32);
#pragma unroll
    for (int ct = 0; ct < 4; ++ct) {
#pragma unroll
      for (int j = 0; j < 8; ++j) acc[ct][j] *= aj[j];
      acc[ct] = wmma_bf16(
          pa, load_b_frag(vB + (size_t)(ct * 16 + n) * SEQ + kb, h), acc[ct]);
    }
  }

  const float inv = 1.0f / l;
  float sj[8];
#pragma unroll
  for (int j = 0; j < 8; ++j) sj[j] = __shfl(inv, 8 * h + j, 32);
#pragma unroll
  for (int ct = 0; ct < 4; ++ct)
#pragma unroll
    for (int j = 0; j < 8; ++j)
      ybuf[(size_t)(bIdx * SEQ + qb + 8 * h + j) * DHEAD + ct * 16 + n] =
          (__bf16)(acc[ct][j] * sj[j]);
}

// ---------------------------------------------------------------------------
// Kernel 3: out = y @ O^T  ([8192,64] x [64,1024] -> [8192,1024] f32).
// One wave per (16-row tile, 256-col group): 2048 waves.
// ---------------------------------------------------------------------------
__global__ void __launch_bounds__(256) oproj(
    const __bf16* __restrict__ ybuf, const __bf16* __restrict__ Ob,
    float* __restrict__ out) {
  const int lane = threadIdx.x & 31;
  const int wid  = blockIdx.x * 8 + (threadIdx.x >> 5);   // 0..2047
  const int rb = (wid >> 2) * 16;
  const int cg = wid & 3;
  const int n = lane & 15, h = lane >> 4;

  const v16bf a0 = load_a_frag(ybuf + (size_t)(rb + n) * DHEAD + 0,  h);
  const v16bf a1 = load_a_frag(ybuf + (size_t)(rb + n) * DHEAD + 32, h);

  for (int ctl = 0; ctl < 16; ++ctl) {
    const int nb = (cg * 16 + ctl) * 16;   // output column tile base
    v8f c = {};
    c = wmma_bf16(a0, load_b_frag(Ob + (size_t)(nb + n) * DHEAD + 0,  h), c);
    c = wmma_bf16(a1, load_b_frag(Ob + (size_t)(nb + n) * DHEAD + 32, h), c);
#pragma unroll
    for (int j = 0; j < 8; ++j)
      out[(size_t)(rb + 8 * h + j) * DMODEL + nb + n] = c[j];
  }
}

// ---------------------------------------------------------------------------
extern "C" void kernel_launch(void* const* d_in, const int* in_sizes, int n_in,
                              void* d_out, int out_size, void* d_ws, size_t ws_size,
                              hipStream_t stream) {
  const float* x = (const float*)d_in[0];
  const float* Q = (const float*)d_in[1];
  const float* K = (const float*)d_in[2];
  const float* V = (const float*)d_in[3];
  const float* O = (const float*)d_in[4];
  float* out = (float*)d_out;

  __bf16* ws = (__bf16*)d_ws;
  // element offsets (bf16): weights 4x64K, buffers 4x512K  -> 4.5 MB total
  __bf16* Qt    = ws;                    // [64][1024]
  __bf16* Kt    = ws + 65536;            // [64][1024]
  __bf16* Vt    = ws + 131072;           // [64][1024]
  __bf16* Ob    = ws + 196608;           // [1024][64]
  __bf16* qbuf  = ws + 262144;           // [8192][64]
  __bf16* kbuf  = ws + 786432;           // [8192][64]
  __bf16* vtbuf = ws + 1310720;          // [4][64][2048]
  __bf16* ybuf  = ws + 1835008;          // [8192][64]

  prep_weights<<<256, 256, 0, stream>>>(Q, K, V, O, Qt, Kt, Vt, Ob);
  proj_qkv<<<64, 256, 0, stream>>>(x, Qt, Kt, Vt, qbuf, kbuf, vtbuf);
  attn_flash<<<64, 256, 0, stream>>>(qbuf, kbuf, vtbuf, ybuf);
  oproj<<<256, 256, 0, stream>>>(ybuf, Ob, out);
}
